// FeatureGroupEncoder_50543175140060
// MI455X (gfx1250) — matrix operand order repaired and checked by reading, compiled, not run on hardware
//
#include <hip/hip_runtime.h>
#include <hip/hip_bf16.h>

// ---------------- problem constants ----------------
#define B_    8
#define S_    1024
#define FIN_  256
#define E_    512
#define INNER_ 1024
#define NH_   4
#define DH_   256
#define KS_   4
#define NB_   256
#define BS_   (B_ * S_)          // 8192 tokens
#define SCALE_ 0.0625f           // DH^-0.5

// ---------------- WMMA types / helpers ----------------
typedef __attribute__((ext_vector_type(16))) __bf16 bf16x16;
typedef __attribute__((ext_vector_type(8)))  float  v8f;

union BFrag {
  bf16x16 v;
  unsigned short u[16];
  unsigned int   d[8];
};

__device__ inline v8f zero8() { v8f z = {0.f,0.f,0.f,0.f,0.f,0.f,0.f,0.f}; return z; }

__device__ inline unsigned short f2bf(float f) {
  unsigned u = __builtin_bit_cast(unsigned, f);
  unsigned r = u + 0x7FFFu + ((u >> 16) & 1u);   // round-to-nearest-even
  return (unsigned short)(r >> 16);
}
__device__ inline float bf2f(unsigned short u) {
  unsigned v = ((unsigned)u) << 16;
  return __builtin_bit_cast(float, v);
}

// A fragment: 16x32 bf16 tile (M x K), row-major src with stride ldk.
// ISA layout: lanes 0-15 -> M=lane, K in {0..7, 16..23}; lanes 16-31 -> K in {8..15, 24..31}.
__device__ inline bf16x16 load_a_frag(const unsigned short* src, int ldk, int lane) {
  BFrag f;
  const unsigned short* p = src + (size_t)(lane & 15) * ldk;
  const int kb = (lane < 16) ? 0 : 8;
#pragma unroll
  for (int v = 0; v < 8; ++v) {
    int k = (v < 4) ? (kb + 2 * v) : (16 + kb + 2 * (v - 4));
    f.d[v] = *(const unsigned int*)(p + k);
  }
  return f.v;
}

// B fragment: 32x16 bf16 tile (K x N) where element B[k][n] = src[n*ldk + k]
// lanes 0-15 -> N=lane, K in {0..7, 16..23} ; lanes 16-31 -> N=lane-16, K in {8..15, 24..31}.
__device__ inline bf16x16 load_b_frag(const unsigned short* src, int ldk, int lane) {
  BFrag f;
  const unsigned short* p = src + (size_t)(lane & 15) * ldk + ((lane >> 4) << 3);
#pragma unroll
  for (int v = 0; v < 8; ++v) f.d[v] = *(const unsigned int*)(p + ((v < 4) ? (2 * v) : (16 + 2 * (v - 4))));
  return f.v;
}

__device__ inline v8f wmma_bf16(bf16x16 a, bf16x16 b, v8f c) {
  return __builtin_amdgcn_wmma_f32_16x16x32_bf16(false, a, false, b, (short)0, c, false, false);
}

// gfx1250 async global->LDS DMA (per-lane 16B chunk), tracked by ASYNCcnt.
__device__ inline void async_copy_b128(unsigned lds_off_bytes, const void* gaddr) {
  asm volatile("global_load_async_to_lds_b128 %0, %1, off"
               :: "v"(lds_off_bytes), "v"(gaddr)
               : "memory");
}
__device__ inline void async_wait0() {
  asm volatile("s_wait_asynccnt 0x0" ::: "memory");
}

// reductions within 16-lane halves (wave32)
__device__ inline float half16_max(float v) {
#pragma unroll
  for (int m = 1; m < 16; m <<= 1) v = fmaxf(v, __shfl_xor(v, m, 32));
  return v;
}
__device__ inline float half16_sum(float v) {
#pragma unroll
  for (int m = 1; m < 16; m <<= 1) v += __shfl_xor(v, m, 32);
  return v;
}

__device__ inline float silu_f(float x) { return x / (1.f + __expf(-x)); }

// ---------------- kernels ----------------

__global__ void cvt_bf16_kernel(const float* __restrict__ src,
                                unsigned short* __restrict__ dst, int n) {
  int i = blockIdx.x * blockDim.x + threadIdx.x;
  if (i < n) dst[i] = f2bf(src[i]);
}

// C[M,N] (f32) = A[M,K] (bf16) @ W[N,K]^T (bf16) + bias
// 8 waves arranged 2(M) x 4(N); each wave computes a 16x64 strip (4 accumulators
// sharing one A fragment -> 4x A reuse). Block tile: 32 x 256.
__global__ __launch_bounds__(256) void gemm_bf16_kernel(
    const unsigned short* __restrict__ A, const unsigned short* __restrict__ W,
    const float* __restrict__ bias, float* __restrict__ C, int M, int N, int K) {
  const int lane = threadIdx.x & 31;
  const int w    = threadIdx.x >> 5;
  const int mw   = w >> 2;         // 0..1
  const int nw   = w & 3;          // 0..3
  const int m0   = blockIdx.y * 32 + mw * 16;
  const int n0   = blockIdx.x * 256 + nw * 64;
  const unsigned short* Ab = A + (size_t)m0 * K;
  const unsigned short* Wb = W + (size_t)n0 * K;
  v8f c0 = zero8(), c1 = zero8(), c2 = zero8(), c3 = zero8();
  for (int k0 = 0; k0 < K; k0 += 32) {
    __builtin_prefetch(Wb + (size_t)(lane & 15) * K + k0 + 256, 0, 1);
    bf16x16 a  = load_a_frag(Ab + k0, K, lane);
    bf16x16 b0 = load_b_frag(Wb + k0, K, lane);
    bf16x16 b1 = load_b_frag(Wb + (size_t)16 * K + k0, K, lane);
    bf16x16 b2 = load_b_frag(Wb + (size_t)32 * K + k0, K, lane);
    bf16x16 b3 = load_b_frag(Wb + (size_t)48 * K + k0, K, lane);
    c0 = wmma_bf16(a, b0, c0);
    c1 = wmma_bf16(a, b1, c1);
    c2 = wmma_bf16(a, b2, c2);
    c3 = wmma_bf16(a, b3, c3);
  }
  const int n  = lane & 15;
  const int mo = (lane >> 4) << 3;
  float bv0 = bias ? bias[n0 + n]      : 0.f;
  float bv1 = bias ? bias[n0 + 16 + n] : 0.f;
  float bv2 = bias ? bias[n0 + 32 + n] : 0.f;
  float bv3 = bias ? bias[n0 + 48 + n] : 0.f;
#pragma unroll
  for (int r = 0; r < 8; ++r) {
    size_t base = (size_t)(m0 + mo + r) * N + n0 + n;
    C[base]      = c0[r] + bv0;
    C[base + 16] = c1[r] + bv1;
    C[base + 32] = c2[r] + bv2;
    C[base + 48] = c3[r] + bv3;
  }
}

// row LayerNorm (f32 in) -> bf16 out, width C (C multiple of 256)
__global__ __launch_bounds__(256) void ln_bf16_kernel(
    const float* __restrict__ in, const float* __restrict__ wgt,
    unsigned short* __restrict__ out, int C) {
  __shared__ float s1[256], s2[256];
  const float* row = in + (size_t)blockIdx.x * C;
  float a = 0.f, b = 0.f;
  for (int c = threadIdx.x; c < C; c += 256) { float v = row[c]; a += v; b += v * v; }
  s1[threadIdx.x] = a; s2[threadIdx.x] = b;
  __syncthreads();
  for (int st = 128; st > 0; st >>= 1) {
    if (threadIdx.x < st) { s1[threadIdx.x] += s1[threadIdx.x + st]; s2[threadIdx.x] += s2[threadIdx.x + st]; }
    __syncthreads();
  }
  float mean = s1[0] / C;
  float var  = s2[0] / C - mean * mean;
  float rstd = rsqrtf(var + 1e-5f);
  for (int c = threadIdx.x; c < C; c += 256)
    out[(size_t)blockIdx.x * C + c] = f2bf((row[c] - mean) * rstd * wgt[c]);
}

// causal depthwise conv (K=4) + SiLU + block-diagonal q/k/v projection.
__global__ __launch_bounds__(256) void conv_qkv_kernel(
    const float* __restrict__ up, const float* __restrict__ conv_w,
    const float* __restrict__ conv_b, const float* __restrict__ Wq,
    const float* __restrict__ Wk, const float* __restrict__ Wv,
    float* __restrict__ xc, unsigned short* __restrict__ qh,
    unsigned short* __restrict__ kh, unsigned short* __restrict__ vh) {
  const int bs = blockIdx.x;
  const int b = bs >> 10, s = bs & (S_ - 1);
  const int nb = threadIdx.x;
  float xm[4], xcv[4];
#pragma unroll
  for (int i = 0; i < 4; ++i) {
    int ch = nb * 4 + i;
    float acc = conv_b[ch];
#pragma unroll
    for (int j = 0; j < KS_; ++j) {
      int sp = s - (KS_ - 1) + j;
      if (sp >= 0) acc += conv_w[ch * KS_ + j] * up[((size_t)(b * S_ + sp)) * (2 * INNER_) + ch];
    }
    xm[i]  = up[(size_t)bs * (2 * INNER_) + ch];
    xcv[i] = silu_f(acc);
    xc[(size_t)bs * INNER_ + ch] = xcv[i];
  }
#pragma unroll
  for (int o = 0; o < 4; ++o) {
    float qv = 0.f, kv = 0.f, vv = 0.f;
#pragma unroll
    for (int i = 0; i < 4; ++i) {
      qv += xcv[i] * Wq[nb * 16 + o * 4 + i];
      kv += xcv[i] * Wk[nb * 16 + o * 4 + i];
      vv += xm[i]  * Wv[nb * 16 + o * 4 + i];
    }
    int ch = nb * 4 + o;
    int h = ch >> 8, dh = ch & 255;
    size_t idx = ((size_t)(b * NH_ + h) * S_ + s) * DH_ + dh;
    qh[idx] = f2bf(qv); kh[idx] = f2bf(kv); vh[idx] = f2bf(vv);
  }
}

// input/forget gates: ig/fg[b,h,s] = concat(q,k,v) . W_row + b
__global__ __launch_bounds__(128) void gates_kernel(
    const unsigned short* __restrict__ qh, const unsigned short* __restrict__ kh,
    const unsigned short* __restrict__ vh, const float* __restrict__ Wig,
    const float* __restrict__ big, const float* __restrict__ Wfg,
    const float* __restrict__ bfg, float* __restrict__ ig, float* __restrict__ fg) {
  const int bs = blockIdx.x;
  const int b = bs >> 10, s = bs & (S_ - 1);
  float pi[4] = {0, 0, 0, 0}, pf[4] = {0, 0, 0, 0};
  for (int j = threadIdx.x; j < 3 * INNER_; j += 128) {
    int part = j >> 10;
    int c = j & (INNER_ - 1);
    int h = c >> 8, dh = c & 255;
    size_t idx = ((size_t)(b * NH_ + h) * S_ + s) * DH_ + dh;
    const unsigned short* src = (part == 0) ? qh : ((part == 1) ? kh : vh);
    float g = bf2f(src[idx]);
#pragma unroll
    for (int t = 0; t < 4; ++t) {
      pi[t] += g * Wig[t * 3 * INNER_ + j];
      pf[t] += g * Wfg[t * 3 * INNER_ + j];
    }
  }
  __shared__ float red[8][128];
#pragma unroll
  for (int t = 0; t < 4; ++t) { red[t][threadIdx.x] = pi[t]; red[4 + t][threadIdx.x] = pf[t]; }
  __syncthreads();
  for (int st = 64; st > 0; st >>= 1) {
    if (threadIdx.x < st)
      for (int t = 0; t < 8; ++t) red[t][threadIdx.x] += red[t][threadIdx.x + st];
    __syncthreads();
  }
  if (threadIdx.x < 4) {
    int h = threadIdx.x;
    ig[((size_t)(b * NH_ + h)) * S_ + s] = red[h][0] + big[h];
    fg[((size_t)(b * NH_ + h)) * S_ + s] = red[4 + h][0] + bfg[h];
  }
}

// per-(b,h) inclusive cumsum of log-sigmoid(fg)
__global__ void scan_kernel(const float* __restrict__ fg, float* __restrict__ acum) {
  int idx = threadIdx.x;
  if (idx >= B_ * NH_) return;
  const float* f = fg + (size_t)idx * S_;
  float* A = acum + (size_t)idx * S_;
  float run = 0.f;
  for (int t = 0; t < S_; ++t) {
    float x = f[t];
    run += fminf(x, 0.f) - log1pf(__expf(-fabsf(x)));
    A[t] = run;
  }
}

// ---- attention LDS layout (single allocation so async LDS offsets are absolute) ----
#define AQ_OFF 0                               // 16x256 bf16  (8192 B)
#define AK_OFF (AQ_OFF + 16 * DH_ * 2)         // 32x256 bf16  (16384 B)
#define AV_OFF (AK_OFF + 32 * DH_ * 2)         // 256x32 bf16 transposed (16384 B)
#define AC_OFF (AV_OFF + 32 * DH_ * 2)         // 4 x 16x32 bf16 (4096 B)
#define ARS_OFF (AC_OFF + 4 * 16 * 32 * 2)     // 4x16 f32
#define ARQ_OFF (ARS_OFF + 4 * 16 * 4)         // 4x16 f32
#define ASMEM_BYTES (ARQ_OFF + 4 * 16 * 4)

// mLSTM attention core + fused per-head LayerNorm.
// grid (S/16, B*NH), 128 threads = 4 waves; wave w owns DH columns [64w,64w+64).
__global__ __launch_bounds__(128) void attn_kernel(
    const unsigned short* __restrict__ qh, const unsigned short* __restrict__ kh,
    const unsigned short* __restrict__ vh, const float* __restrict__ acum,
    const float* __restrict__ ig, const float* __restrict__ mhln_w,
    float* __restrict__ hn) {
  __shared__ alignas(128) unsigned char smem[ASMEM_BYTES];
  unsigned short* Qs = (unsigned short*)(smem + AQ_OFF);
  unsigned short* Ks = (unsigned short*)(smem + AK_OFF);
  unsigned short* Vt = (unsigned short*)(smem + AV_OFF);
  unsigned short* Cs = (unsigned short*)(smem + AC_OFF);
  float* red_s = (float*)(smem + ARS_OFF);
  float* red_q = (float*)(smem + ARQ_OFF);

  const int qi = blockIdx.x;
  const int bh = blockIdx.y;
  const int b = bh / NH_, h = bh % NH_;
  const int tid = threadIdx.x;
  const int lane = tid & 31;
  const int w = tid >> 5;
  const int ln = lane & 15;
  const int mofs = (lane >> 4) << 3;

  const unsigned short* qb = qh + (size_t)bh * S_ * DH_;
  const unsigned short* kb = kh + (size_t)bh * S_ * DH_;
  const unsigned short* vb = vh + (size_t)bh * S_ * DH_;
  const float* Ab = acum + (size_t)bh * S_;
  const float* Ib = ig + (size_t)bh * S_;

  // async DMA of the Q tile (16x256 bf16 = 512 x 16B chunks, 4 per thread)
#pragma unroll
  for (int c = 0; c < 4; ++c) {
    int ci = tid + c * 128;
    async_copy_b128((unsigned)(AQ_OFF + ci * 16),
                    qb + (((size_t)(qi * 16 + (ci >> 5))) << 8) + (ci & 31) * 8);
  }

  float m[8], csum[8], a_s[8];
  v8f acc[4];
#pragma unroll
  for (int r = 0; r < 8; ++r) {
    m[r] = -3.0e38f; csum[r] = 0.f;
    a_s[r] = Ab[qi * 16 + mofs + r];
  }
#pragma unroll
  for (int j = 0; j < 4; ++j) acc[j] = zero8();

  const int tbmax = (qi * 16 + 15) >> 5;
  for (int tb = 0; tb <= tbmax; ++tb) {
    __syncthreads();   // previous iteration finished reading Ks/Vt
    // async DMA of K tile (32x256 bf16 = 1024 x 16B chunks, 8 per thread)
#pragma unroll
    for (int c = 0; c < 8; ++c) {
      int ci = tid + c * 128;
      async_copy_b128((unsigned)(AK_OFF + ci * 16),
                      kb + (((size_t)(tb * 32 + (ci >> 5))) << 8) + (ci & 31) * 8);
    }
    // V tile stored transposed [dh][t] so PV B-fragments are LDS-contiguous
    for (int i = tid; i < 32 * DH_; i += 128) {
      int tl = i >> 8, dh = i & 255;
      Vt[dh * 32 + tl] = vb[(size_t)(tb * 32 + tl) * DH_ + dh];
    }
    async_wait0();
    __syncthreads();

    // scores: q . k for 16 queries x 32 keys
    v8f sc0 = zero8(), sc1 = zero8();
    for (int kc = 0; kc < DH_ / 32; ++kc) {
      bf16x16 af = load_a_frag(Qs + kc * 32, DH_, lane);
      bf16x16 b0 = load_b_frag(Ks + kc * 32, DH_, lane);
      bf16x16 b1 = load_b_frag(Ks + 16 * DH_ + kc * 32, DH_, lane);
      sc0 = wmma_bf16(af, b0, sc0);
      sc1 = wmma_bf16(af, b1, sc1);
    }

    const int t0 = tb * 32 + ln;
    const int t1 = t0 + 16;
    const float at0 = Ab[t0], it0 = Ib[t0];
    const float at1 = Ab[t1], it1 = Ib[t1];

#pragma unroll
    for (int r = 0; r < 8; ++r) {
      int srow = qi * 16 + mofs + r;
      float ld0 = (srow >= t0) ? (a_s[r] - at0 + it0) : -1e30f;
      float ld1 = (srow >= t1) ? (a_s[r] - at1 + it1) : -1e30f;
      float tmax = half16_max(fmaxf(ld0, ld1));
      float mn = fmaxf(m[r], tmax);
      float rs = __expf(m[r] - mn);
      float c0 = sc0[r] * SCALE_ * __expf(ld0 - mn);
      float c1 = sc1[r] * SCALE_ * __expf(ld1 - mn);
      csum[r] = csum[r] * rs + half16_sum(c0 + c1);
      m[r] = mn;
#pragma unroll
      for (int j = 0; j < 4; ++j) acc[j][r] *= rs;
      Cs[w * 16 * 32 + (mofs + r) * 32 + ln] = f2bf(c0);
      Cs[w * 16 * 32 + (mofs + r) * 32 + 16 + ln] = f2bf(c1);
    }

    // acc += C(16x32) @ V(32 x 64-chunk)
    bf16x16 afC = load_a_frag(Cs + w * 16 * 32, 32, lane);
#pragma unroll
    for (int j = 0; j < 4; ++j) {
      bf16x16 bf = load_b_frag(Vt + (size_t)(w * 64 + j * 16) * 32, 32, lane);
      acc[j] = wmma_bf16(afC, bf, acc[j]);
    }
  }

  // normalize: n = max(|csum|, exp(-m)); hh = acc / (n + 1e-6)
  float psum[8], psq[8];
#pragma unroll
  for (int r = 0; r < 8; ++r) {
    float nv = fmaxf(fabsf(csum[r]), __expf(-m[r]));
    float inv = 1.f / (nv + 1e-6f);
    float s = 0.f, q = 0.f;
#pragma unroll
    for (int j = 0; j < 4; ++j) {
      float v = acc[j][r] * inv;
      acc[j][r] = v;
      s += v; q += v * v;
    }
    psum[r] = half16_sum(s);
    psq[r] = half16_sum(q);
  }
  if (ln == 0) {
#pragma unroll
    for (int r = 0; r < 8; ++r) { red_s[w * 16 + mofs + r] = psum[r]; red_q[w * 16 + mofs + r] = psq[r]; }
  }
  __syncthreads();

  // fused multi-head LayerNorm over DH, then scatter into (b,s,inner)
#pragma unroll
  for (int r = 0; r < 8; ++r) {
    int row = mofs + r;
    float ts = red_s[row] + red_s[16 + row] + red_s[32 + row] + red_s[48 + row];
    float tq = red_q[row] + red_q[16 + row] + red_q[32 + row] + red_q[48 + row];
    float mean = ts * (1.f / DH_);
    float var  = tq * (1.f / DH_) - mean * mean;
    float rstd = rsqrtf(var + 1e-5f);
    int srow = qi * 16 + row;
#pragma unroll
    for (int j = 0; j < 4; ++j) {
      int dh = w * 64 + j * 16 + ln;
      float v = (acc[j][r] - mean) * rstd * mhln_w[h * DH_ + dh];
      hn[((size_t)(b * S_ + srow)) * INNER_ + h * DH_ + dh] = v;
    }
  }
}

// h_state = (hn + skip*xc) * silu(z) -> bf16
__global__ __launch_bounds__(256) void hstate_kernel(
    const float* __restrict__ hn, const float* __restrict__ xc,
    const float* __restrict__ up, const float* __restrict__ skip,
    unsigned short* __restrict__ hs) {
  size_t i = (size_t)blockIdx.x * 256 + threadIdx.x;
  int c = (int)(i & (INNER_ - 1));
  size_t bs = i >> 10;
  float z = up[bs * (2 * INNER_) + INNER_ + c];
  float v = (hn[i] + skip[c] * xc[i]) * silu_f(z);
  hs[i] = f2bf(v);
}

// out = LN(xp + y) * w  (C = 512)
__global__ __launch_bounds__(256) void addln_kernel(
    const float* __restrict__ xp, const float* __restrict__ y,
    const float* __restrict__ w, float* __restrict__ out) {
  __shared__ float s1[256], s2[256];
  size_t base = (size_t)blockIdx.x * E_;
  float v0 = xp[base + threadIdx.x] + y[base + threadIdx.x];
  float v1 = xp[base + 256 + threadIdx.x] + y[base + 256 + threadIdx.x];
  s1[threadIdx.x] = v0 + v1;
  s2[threadIdx.x] = v0 * v0 + v1 * v1;
  __syncthreads();
  for (int st = 128; st > 0; st >>= 1) {
    if (threadIdx.x < st) { s1[threadIdx.x] += s1[threadIdx.x + st]; s2[threadIdx.x] += s2[threadIdx.x + st]; }
    __syncthreads();
  }
  float mean = s1[0] * (1.f / E_);
  float var  = s2[0] * (1.f / E_) - mean * mean;
  float rstd = rsqrtf(var + 1e-5f);
  out[base + threadIdx.x] = (v0 - mean) * rstd * w[threadIdx.x];
  out[base + 256 + threadIdx.x] = (v1 - mean) * rstd * w[256 + threadIdx.x];
}

// ---------------- host launcher ----------------
extern "C" void kernel_launch(void* const* d_in, const int* in_sizes, int n_in,
                              void* d_out, int out_size, void* d_ws, size_t ws_size,
                              hipStream_t stream) {
  const float* x       = (const float*)d_in[0];
  const float* W_in    = (const float*)d_in[1];
  const float* b_in    = (const float*)d_in[2];
  const float* ln1_w   = (const float*)d_in[3];
  const float* W_up    = (const float*)d_in[4];
  const float* conv_w  = (const float*)d_in[5];
  const float* conv_b  = (const float*)d_in[6];
  const float* Wq      = (const float*)d_in[7];
  const float* Wk      = (const float*)d_in[8];
  const float* Wv      = (const float*)d_in[9];
  const float* W_ig    = (const float*)d_in[10];
  const float* b_ig    = (const float*)d_in[11];
  const float* W_fg    = (const float*)d_in[12];
  const float* b_fg    = (const float*)d_in[13];
  const float* mhln_w  = (const float*)d_in[14];
  const float* skip    = (const float*)d_in[15];
  const float* W_down  = (const float*)d_in[16];
  const float* ln_post = (const float*)d_in[17];

  char* ws = (char*)d_ws;
  size_t off = 0;
  auto take = [&](size_t bytes) -> void* {
    void* p = ws + off;
    off += (bytes + 255) & ~(size_t)255;
    return p;
  };

  unsigned short* xb   = (unsigned short*)take((size_t)BS_ * FIN_ * 2);
  unsigned short* Winb = (unsigned short*)take((size_t)E_ * FIN_ * 2);
  unsigned short* Wupb = (unsigned short*)take((size_t)2 * INNER_ * E_ * 2);
  unsigned short* Wdnb = (unsigned short*)take((size_t)E_ * INNER_ * 2);
  float*          xp   = (float*)take((size_t)BS_ * E_ * 4);
  unsigned short* hb   = (unsigned short*)take((size_t)BS_ * E_ * 2);
  float*          up   = (float*)take((size_t)BS_ * 2 * INNER_ * 4);
  float*          xc   = (float*)take((size_t)BS_ * INNER_ * 4);
  unsigned short* qhb  = (unsigned short*)take((size_t)BS_ * INNER_ * 2);
  unsigned short* khb  = (unsigned short*)take((size_t)BS_ * INNER_ * 2);
  unsigned short* vhb  = (unsigned short*)take((size_t)BS_ * INNER_ * 2);
  float*          igv  = (float*)take((size_t)B_ * NH_ * S_ * 4);
  float*          fgv  = (float*)take((size_t)B_ * NH_ * S_ * 4);
  float*          acum = (float*)take((size_t)B_ * NH_ * S_ * 4);
  float*          hn   = (float*)take((size_t)BS_ * INNER_ * 4);
  unsigned short* hs   = (unsigned short*)take((size_t)BS_ * INNER_ * 2);
  float*          yv   = (float*)take((size_t)BS_ * E_ * 4);

  // 1. convert inputs/weights to bf16
  {
    int n;
    n = BS_ * FIN_;        cvt_bf16_kernel<<<(n + 255) / 256, 256, 0, stream>>>(x, xb, n);
    n = E_ * FIN_;         cvt_bf16_kernel<<<(n + 255) / 256, 256, 0, stream>>>(W_in, Winb, n);
    n = 2 * INNER_ * E_;   cvt_bf16_kernel<<<(n + 255) / 256, 256, 0, stream>>>(W_up, Wupb, n);
    n = E_ * INNER_;       cvt_bf16_kernel<<<(n + 255) / 256, 256, 0, stream>>>(W_down, Wdnb, n);
  }

  // 2. xp = x @ W_in^T + b_in
  gemm_bf16_kernel<<<dim3(E_ / 256, BS_ / 32), 256, 0, stream>>>(
      xb, Winb, b_in, xp, BS_, E_, FIN_);

  // 3. h = LN(xp) * ln1_w (bf16)
  ln_bf16_kernel<<<BS_, 256, 0, stream>>>(xp, ln1_w, hb, E_);

  // 4. up = h @ W_up^T
  gemm_bf16_kernel<<<dim3(2 * INNER_ / 256, BS_ / 32), 256, 0, stream>>>(
      hb, Wupb, nullptr, up, BS_, 2 * INNER_, E_);

  // 5. conv + silu + q/k/v
  conv_qkv_kernel<<<BS_, NB_, 0, stream>>>(up, conv_w, conv_b, Wq, Wk, Wv,
                                           xc, qhb, khb, vhb);

  // 6. gates
  gates_kernel<<<BS_, 128, 0, stream>>>(qhb, khb, vhb, W_ig, b_ig, W_fg, b_fg,
                                        igv, fgv);

  // 7. cumulative log-sigmoid scan
  scan_kernel<<<1, 32, 0, stream>>>(fgv, acum);

  // 8. mLSTM attention + per-head LN
  attn_kernel<<<dim3(S_ / 16, B_ * NH_), 128, 0, stream>>>(
      qhb, khb, vhb, acum, igv, mhln_w, hn);

  // 9. h_state = (hn + skip*xc) * silu(z)
  hstate_kernel<<<(BS_ * INNER_) / 256, 256, 0, stream>>>(hn, xc, up, skip, hs);

  // 10. y = h_state @ W_down^T
  gemm_bf16_kernel<<<dim3(E_ / 256, BS_ / 32), 256, 0, stream>>>(
      hs, Wdnb, nullptr, yv, BS_, E_, INNER_);

  // 11. out = LN(xp + y) * ln_post_w
  addln_kernel<<<BS_, 256, 0, stream>>>(xp, yv, ln_post, (float*)d_out);
}
